// CRFLayer_4853313044523
// MI455X (gfx1250) — matrix operand order repaired and checked by reading, not compiled, not run on hardware
//
#include <hip/hip_runtime.h>

typedef __attribute__((ext_vector_type(16))) _Float16 v16h;
typedef __attribute__((ext_vector_type(8)))  _Float16 v8h;
typedef __attribute__((ext_vector_type(8)))  float    v8f;
typedef __attribute__((ext_vector_type(4)))  float    v4f;

static constexpr int BB = 512;   // batch
static constexpr int SS = 1024;  // sequence length
static constexpr int TT = 64;    // tags / states
static constexpr int AROW = 72;  // padded LDS row stride (halves): 144B -> conflict-free b128 loads
static constexpr int EROW = 72;
static constexpr int PFD  = 8;   // prefetch distance (steps); OOB prefetch is dropped by HW

// Raw hardware transcendentals (v_exp_f32 / v_log_f32 are base-2).
// Safe here: log args are >= 0.9 (no denorm/zero path needed), exp args <= 0.
__device__ __forceinline__ float fast_exp(float x) {
    return __builtin_amdgcn_exp2f(x * 1.4426950408889634f);
}
__device__ __forceinline__ float fast_log(float x) {
    return 0.6931471805599453f * __builtin_amdgcn_logf(x);
}

// VALU cross-lane reduction within each 16-lane DPP row (wave32 = 2 rows,
// matching the two lane-halves of the WMMA C/D layout). Avoids ds_bpermute
// and its DScnt serialization against the LDS A-matrix staging.
template <int CTRL>
__device__ __forceinline__ float dpp_rot(float v) {
    return __int_as_float(__builtin_amdgcn_update_dpp(
        0, __float_as_int(v), CTRL, 0xf, 0xf, true));
}
__device__ __forceinline__ float rowmax16(float m) {
    m = fmaxf(m, dpp_rot<0x121>(m));  // row_ror:1
    m = fmaxf(m, dpp_rot<0x122>(m));  // row_ror:2
    m = fmaxf(m, dpp_rot<0x124>(m));  // row_ror:4
    m = fmaxf(m, dpp_rot<0x128>(m));  // row_ror:8
    return m;
}
__device__ __forceinline__ float rowsum16(float s) {
    s += dpp_rot<0x121>(s);
    s += dpp_rot<0x122>(s);
    s += dpp_rot<0x124>(s);
    s += dpp_rot<0x128>(s);
    return s;
}
__device__ __forceinline__ float max4(v4f v) {
    return fmaxf(fmaxf(v.x, v.y), fmaxf(v.z, v.w));
}

// ---------------------------------------------------------------------------
// Forward (log-partition) kernel.
// Grid: B/16 workgroups; Block: 128 threads = 4 wave32 (one 16-state N-tile each).
// Recurrence per step (exp-rescaled so the T x T logsumexp becomes a GEMM):
//   m      = rowmax(alpha)                    (DPP + one b128 LDS combine)
//   a      = exp(alpha - m)  in (0,1]  -> f16 in LDS
//   v      = a * E           E = exp(transitions), constant, f16 in registers (B frags)
//   alpha' = m + log(v) + emit_t ; masked update (branchless)
// Emissions/mask loads are software-pipelined one step ahead + speculative
// prefetch PFD steps ahead, so the sequential chain never stalls on HBM.
// ---------------------------------------------------------------------------
__global__ __launch_bounds__(128)
void crf_forward_kernel(const float* __restrict__ emissions,
                        const float* __restrict__ mask,
                        const float* __restrict__ startT,
                        const float* __restrict__ endT,
                        const float* __restrict__ trans,
                        float* __restrict__ denPart)
{
    __shared__ __align__(16) _Float16 eSh[TT * EROW];   // E = exp(transitions), f16
    __shared__ __align__(16) _Float16 aSh[16 * AROW];   // rescaled alpha (A matrix), f16
    __shared__ __align__(16) float wmaxSh[16][4];       // [row][wave] -> one b128 read
    __shared__ __align__(16) float wsumSh[16][4];

    const int tid  = threadIdx.x;
    const int w    = tid >> 5;      // wave id (N-tile)
    const int lane = tid & 31;
    const int half = lane >> 4;     // 0: lanes 0-15, 1: lanes 16-31
    const int nIdx = lane & 15;
    const int col  = w * 16 + nIdx; // state j owned by this lane in C/D layout
    const int b0   = blockIdx.x * 16;

    // ---- prologue: E = exp(transitions) into LDS as f16 ----
    for (int idx = tid; idx < TT * TT; idx += 128) {
        int i = idx >> 6, j = idx & 63;
        eSh[i * EROW + j] = (_Float16)fast_exp(trans[idx]);
    }
    __syncthreads();

    // ---- constant B fragments (16-bit B 32x16 layout: lane=N, K groups per half) ----
    v16h Bfrag[2];
    for (int f = 0; f < 2; ++f) {
        union { v16h v; _Float16 h[16]; } u;
        #pragma unroll
        for (int e = 0; e < 16; ++e) {
            int k = f * 32 + (e < 8 ? (half * 8 + e) : (16 + half * 8 + (e - 8)));
            u.h[e] = eSh[k * EROW + col];
        }
        Bfrag[f] = u.v;
    }

    // Base pointers; per-row offsets (r * 256KB / r * 4KB) fold into the 24-bit
    // instruction offset, keeping VGPR/SALU addressing overhead minimal.
    const float* ebase = emissions + (size_t)(b0 + half * 8) * (SS * TT) + col;
    const float* mbase = mask      + (size_t)(b0 + half * 8) * SS;

    // ---- init alpha (C/D layout: VGPR r -> row r + 8*half, col = w*16 + nIdx) ----
    const float stc = startT[col];
    const float etc = endT[col];
    float alpha[8];
    #pragma unroll
    for (int r = 0; r < 8; ++r)
        alpha[r] = stc + ebase[(size_t)r * (SS * TT)];

    // preload step t=1 (consumed in iteration t=1, next batch loaded there)
    float emit[8], mk[8];
    #pragma unroll
    for (int r = 0; r < 8; ++r) {
        emit[r] = ebase[(size_t)r * (SS * TT) + 1 * TT];
        mk[r]   = mbase[(size_t)r * SS + 1];
    }

    for (int t = 1; t < SS; ++t) {
        // pipeline: issue loads for t+1 now, consume them next iteration
        const int tn = (t + 1 < SS) ? (t + 1) : (SS - 1);
        float emitN[8], mkN[8];
        #pragma unroll
        for (int r = 0; r < 8; ++r) {
            emitN[r] = ebase[(size_t)r * (SS * TT) + (size_t)tn * TT];
            mkN[r]   = mbase[(size_t)r * SS + tn];
            // speculative prefetch PFD steps ahead (OOB silently dropped)
            __builtin_prefetch(ebase + (size_t)r * (SS * TT) + (size_t)(t + PFD) * TT, 0, 3);
        }

        // phase 1: per-wave row max over this wave's 16 states (DPP row reduce)
        #pragma unroll
        for (int r = 0; r < 8; ++r) {
            float m = rowmax16(alpha[r]);
            if (nIdx == 0) wmaxSh[r + half * 8][w] = m;
        }
        __syncthreads();

        // phase 2: combine maxes (1 x b128 per row), a = exp(alpha - m) -> f16 LDS
        float mrow[8];
        #pragma unroll
        for (int r = 0; r < 8; ++r) {
            int row = r + half * 8;
            float m = max4(*reinterpret_cast<const v4f*>(&wmaxSh[row][0]));
            mrow[r] = m;
            aSh[row * AROW + col] = (_Float16)fast_exp(alpha[r] - m);
        }
        __syncthreads();

        // phase 3: v = a * E via 2 chained WMMAs (K = 0..31, 32..63)
        // A 16x32 f16 layout: lane = row (nIdx), K offsets {0..7,16..23} / {8..15,24..31}
        v8f acc = {};
        #pragma unroll
        for (int f = 0; f < 2; ++f) {
            int base = nIdx * AROW + f * 32 + (half ? 8 : 0);
            v8h lo = *reinterpret_cast<const v8h*>(&aSh[base]);
            v8h hi = *reinterpret_cast<const v8h*>(&aSh[base + 16]);
            v16h A = __builtin_shufflevector(lo, hi,
                       0,1,2,3,4,5,6,7,8,9,10,11,12,13,14,15);
            acc = __builtin_amdgcn_wmma_f32_16x16x32_f16(
                      false, A, false, Bfrag[f], (short)0, acc, false, false);
        }

        // phase 4: alpha' = m + log(v) + emit, branchless masked select
        #pragma unroll
        for (int r = 0; r < 8; ++r) {
            float nv = mrow[r] + fast_log(acc[r]) + emit[r];
            alpha[r] = (mk[r] > 0.0f) ? nv : alpha[r];
            emit[r] = emitN[r];
            mk[r]   = mkN[r];
        }
    }

    // ---- epilogue: logsumexp_j(alpha + endT), summed over the 16 batch rows ----
    #pragma unroll
    for (int r = 0; r < 8; ++r) alpha[r] += etc;
    #pragma unroll
    for (int r = 0; r < 8; ++r) {
        float m = rowmax16(alpha[r]);
        if (nIdx == 0) wmaxSh[r + half * 8][w] = m;
    }
    __syncthreads();
    #pragma unroll
    for (int r = 0; r < 8; ++r) {
        int row = r + half * 8;
        float m = max4(*reinterpret_cast<const v4f*>(&wmaxSh[row][0]));
        float s = rowsum16(fast_exp(alpha[r] - m));
        if (nIdx == 0) wsumSh[row][w] = s;
    }
    __syncthreads();
    if (tid == 0) {
        float total = 0.0f;
        for (int row = 0; row < 16; ++row) {
            float m = max4(*reinterpret_cast<const v4f*>(&wmaxSh[row][0]));
            v4f sv  = *reinterpret_cast<const v4f*>(&wsumSh[row][0]);
            float s = (sv.x + sv.y) + (sv.z + sv.w);
            total += m + fast_log(s);
        }
        denPart[blockIdx.x] = total;
    }
}

// ---------------------------------------------------------------------------
// Gold-score kernel: one block per batch row, fixed-order tree reduce.
// ---------------------------------------------------------------------------
__global__ __launch_bounds__(256)
void crf_gold_kernel(const float* __restrict__ emissions,
                     const float* __restrict__ mask,
                     const float* __restrict__ startT,
                     const float* __restrict__ endT,
                     const float* __restrict__ trans,
                     const int*   __restrict__ tags,
                     float* __restrict__ goldPart)
{
    __shared__ float red[256];
    __shared__ float redm[256];
    const int b = blockIdx.x, tid = threadIdx.x;
    const int*   tg = tags + (size_t)b * SS;
    const float* mk = mask + (size_t)b * SS;
    const float* em = emissions + (size_t)b * SS * TT;

    float s = 0.0f, ms = 0.0f;
    for (int t = tid; t < SS; t += 256) ms += mk[t];
    for (int t = tid + 1; t < SS; t += 256) {
        int tc = tg[t], tp = tg[t - 1];
        s += (em[t * TT + tc] + trans[tp * TT + tc]) * mk[t];
    }
    red[tid] = s; redm[tid] = ms;
    __syncthreads();
    for (int off = 128; off > 0; off >>= 1) {
        if (tid < off) { red[tid] += red[tid + off]; redm[tid] += redm[tid + off]; }
        __syncthreads();
    }
    if (tid == 0) {
        int t0 = tg[0];
        float sc = red[0] + startT[t0] + em[t0];
        int li = (int)(redm[0] + 0.5f) - 1;
        li = li < 0 ? 0 : (li > SS - 1 ? SS - 1 : li);
        sc += endT[tg[li]];
        goldPart[b] = sc;
    }
}

// ---------------------------------------------------------------------------
// Final deterministic combine: out = sum(gold) - sum(den).
// ---------------------------------------------------------------------------
__global__ void crf_final_kernel(const float* __restrict__ goldPart,
                                 const float* __restrict__ denPart,
                                 float* __restrict__ out)
{
    if (threadIdx.x == 0 && blockIdx.x == 0) {
        float g = 0.0f, d = 0.0f;
        for (int i = 0; i < BB; ++i)      g += goldPart[i];
        for (int i = 0; i < BB / 16; ++i) d += denPart[i];
        out[0] = g - d;
    }
}

extern "C" void kernel_launch(void* const* d_in, const int* in_sizes, int n_in,
                              void* d_out, int out_size, void* d_ws, size_t ws_size,
                              hipStream_t stream) {
    const float* emissions = (const float*)d_in[0];
    const float* mask      = (const float*)d_in[1];
    const float* startT    = (const float*)d_in[2];
    const float* endT      = (const float*)d_in[3];
    const float* trans     = (const float*)d_in[4];
    const int*   tags      = (const int*)d_in[5];
    float* out = (float*)d_out;

    float* goldPart = (float*)d_ws;            // [0 .. 511]
    float* denPart  = goldPart + BB;           // [512 .. 543]

    crf_gold_kernel<<<BB, 256, 0, stream>>>(emissions, mask, startT, endT, trans,
                                            tags, goldPart);
    crf_forward_kernel<<<BB / 16, 128, 0, stream>>>(emissions, mask, startT, endT,
                                                    trans, denPart);
    crf_final_kernel<<<1, 32, 0, stream>>>(goldPart, denPart, out);
}
